// AdaptiveMiddleFusion_57810259804586
// MI455X (gfx1250) — matrix operand order repaired and checked by reading, not compiled, not run on hardware
//
#include <hip/hip_runtime.h>
#include <hip/hip_bf16.h>

typedef __attribute__((ext_vector_type(16))) _Float16 v16h;
typedef __attribute__((ext_vector_type(8)))  float    v8f;

#define LN_EPS 1e-5f
#define WAVES_PER_BLOCK 4
#define TILES_PER_WAVE  8
#define SP 68   // padded LDS row stride in floats (272 B, 16B aligned, conflict-free)

__device__ __forceinline__ float sigmoidf_(float x) {
  return 1.0f / (1.0f + __expf(-x));
}

// ---------------- text pipeline kernel: one 64-thread block per text row ----------------
// Writes text_t = relu(x@W1+b1)@W2+b2, tgate = text_t@Wg_t + bg, quality = sigmoid(||x||-thr)
__global__ void text_kernel(const float* __restrict__ text_feat,
                            const float* __restrict__ W1, const float* __restrict__ b1,
                            const float* __restrict__ W2, const float* __restrict__ b2,
                            const float* __restrict__ Wg, const float* __restrict__ bg,
                            const float* __restrict__ qthr,
                            float* __restrict__ text_t, float* __restrict__ tgate,
                            float* __restrict__ quality, int batch)
{
  __shared__ float s_x[64];
  __shared__ float s_h[128];
  __shared__ float s_t[64];
  const int r = blockIdx.x;
  if (r >= batch) return;
  const int t = threadIdx.x;
  s_x[t] = text_feat[r * 64 + t];
  __syncthreads();
  for (int j = t; j < 128; j += 64) {          // h = relu(x @ W1 + b1), W1 (64,128)
    float acc = b1[j];
#pragma unroll 8
    for (int k = 0; k < 64; ++k) acc = fmaf(s_x[k], W1[k * 128 + j], acc);
    s_h[j] = fmaxf(acc, 0.0f);
  }
  __syncthreads();
  {                                            // text_t = h @ W2 + b2, W2 (128,64)
    float acc = b2[t];
#pragma unroll 8
    for (int k = 0; k < 128; ++k) acc = fmaf(s_h[k], W2[k * 64 + t], acc);
    s_t[t] = acc;
    text_t[r * 64 + t] = acc;
  }
  __syncthreads();
  {                                            // tgate = text_t @ Wg_t + bg, Wg_t = Wg[64:128]
    float acc = bg[t];
#pragma unroll 8
    for (int k = 0; k < 64; ++k) acc = fmaf(s_t[k], Wg[(64 + k) * 64 + t], acc);
    tgate[r * 64 + t] = acc;
  }
  if (t == 0) {
    float s = 0.0f;
    for (int k = 0; k < 64; ++k) s = fmaf(s_x[k], s_x[k], s);
    quality[r] = sigmoidf_(sqrtf(s) - qthr[0]);
  }
}

// ---------------- fused node kernel: one wave per 16-node tile ----------------
struct __align__(16) WaveShm {
  float node[16][SP];   // staged node rows, f32
  float tb[16][SP];     // gathered text_t rows, f32
  float tg[16][SP];     // gathered tgate rows; overwritten with 'enhanced'
  int   seg[16];
  float qb[16];
  float mu[16];
  float rs[16];
};

__global__ void __launch_bounds__(WAVES_PER_BLOCK * 32)
fusion_kernel(const float* __restrict__ node_feat,
              const int*   __restrict__ seg_ids,
              const float* __restrict__ Wg,        // (128,64), rows 0..63 = Wg_n
              const float* __restrict__ text_t,    // (batch,64)
              const float* __restrict__ tgate,     // (batch,64), includes bg
              const float* __restrict__ quality,   // (batch)
              const float* __restrict__ gamma_,
              const float* __restrict__ beta_,
              float* __restrict__ out,
              int total_nodes)
{
  __shared__ WaveShm shm[WAVES_PER_BLOCK];
  const int lane = threadIdx.x & 31;
  const int wid  = threadIdx.x >> 5;
  WaveShm& S = shm[wid];
  const int gwave  = blockIdx.x * WAVES_PER_BLOCK + wid;
  const int ntiles = (total_nodes + 15) >> 4;

  // ---- B fragments for Wg_n, held in registers for the whole wave.
  // B (32x16) k order used here (contiguous, matches A): lanes 0-15 hold K = kt*32 + 0..15 for column N = lane%16
  // (2 values per VGPR, low half first), lanes 16-31 hold K = kt*32 + 16..31.
  v16h bfrag[4][2];
  {
    const int n     = lane & 15;
    const int khalf = (lane >> 4) * 16;
#pragma unroll
    for (int nt = 0; nt < 4; ++nt)
#pragma unroll
      for (int kt = 0; kt < 2; ++kt) {
        const int kb = kt * 32 + khalf;
#pragma unroll
        for (int e = 0; e < 16; ++e)
          bfrag[nt][kt][e] = (_Float16)Wg[(kb + e) * 64 + (nt * 16 + n)];
      }
  }

  for (int i = 0; i < TILES_PER_WAVE; ++i) {
    const int tile = gwave * TILES_PER_WAVE + i;
    if (tile >= ntiles) break;
    const int row0 = tile << 4;

    if (lane < 16) {                     // segment ids + quality for 16 rows
      int r = row0 + lane;
      if (r >= total_nodes) r = total_nodes - 1;
      const int s = seg_ids[r];
      S.seg[lane] = s;
      S.qb[lane]  = quality[s];
    }
    for (int c = lane; c < 256; c += 32) {   // stage node rows (coalesced float4)
      const int m = c >> 4, q = c & 15;
      int r = row0 + m; if (r >= total_nodes) r = total_nodes - 1;
      *(float4*)&S.node[m][q * 4] = ((const float4*)node_feat)[(size_t)r * 16 + q];
    }
    for (int c = lane; c < 256; c += 32) {   // gather text_t / tgate rows (L2 hits)
      const int m = c >> 4, q = c & 15;
      const int s = S.seg[m];
      *(float4*)&S.tb[m][q * 4] = ((const float4*)text_t)[(size_t)s * 16 + q];
      *(float4*)&S.tg[m][q * 4] = ((const float4*)tgate)[(size_t)s * 16 + q];
    }

    // ---- A fragments (node rows, f32 -> f16).
    // A (16x32) k order used here (contiguous, matches B): lane row m = lane%16, h = (lane/16)*16:
    //   elements 0..7  -> K = kt*32 + h + 0..7
    //   elements 8..15 -> K = kt*32 + h + 8 + 0..7
    v16h afrag[2];
    {
      const int m = lane & 15;
      const int h = (lane >> 4) * 16;
      const float* rowp = &S.node[m][0];
#pragma unroll
      for (int kt = 0; kt < 2; ++kt) {
#pragma unroll
        for (int e = 0; e < 8; ++e)
          afrag[kt][e]     = (_Float16)rowp[kt * 32 + h + e];
#pragma unroll
        for (int e = 0; e < 8; ++e)
          afrag[kt][8 + e] = (_Float16)rowp[kt * 32 + h + 8 + e];
      }
    }

    // ---- acc[nt] = node_tile @ Wg_n  (4 N-tiles x 2 K-steps = 8 WMMAs)
    v8f acc[4];
#pragma unroll
    for (int nt = 0; nt < 4; ++nt) {
      v8f c = {};
      c = __builtin_amdgcn_wmma_f32_16x16x32_f16(false, afrag[0], false, bfrag[nt][0],
                                                 (short)0, c, false, false);
      c = __builtin_amdgcn_wmma_f32_16x16x32_f16(false, afrag[1], false, bfrag[nt][1],
                                                 (short)0, c, false, false);
      acc[nt] = c;
    }

    // ---- gate + enhanced at C layout: element (m = v + 8*(lane/16), n = nt*16 + lane%16)
    {
      const int nlo = lane & 15;
      const int mhi = (lane >> 4) * 8;
#pragma unroll
      for (int nt = 0; nt < 4; ++nt)
#pragma unroll
        for (int v = 0; v < 8; ++v) {
          const int m = mhi + v;
          const int n = nt * 16 + nlo;
          const float pre = acc[nt][v] + S.tg[m][n];       // node@Wg_n + tb@Wg_t + bg
          const float g   = sigmoidf_(pre);
          const float e   = S.node[m][n] + S.qb[m] * g * S.tb[m][n];
          S.tg[m][n] = e;                                  // now holds 'enhanced'
        }
    }

    // ---- LayerNorm stats per row (both half-waves compute row lane%16; same-addr LDS broadcast)
    {
      const int m = lane & 15;
      float s = 0.0f, s2 = 0.0f;
#pragma unroll
      for (int q = 0; q < 16; ++q) {
        const float4 v = *(const float4*)&S.tg[m][q * 4];
        s  += v.x + v.y + v.z + v.w;
        s2 += v.x * v.x + v.y * v.y + v.z * v.z + v.w * v.w;
      }
      const float mu  = s * (1.0f / 64.0f);
      const float var = s2 * (1.0f / 64.0f) - mu * mu;
      const float rs  = __builtin_amdgcn_rsqf(var + LN_EPS);
      if (lane < 16) { S.mu[m] = mu; S.rs[m] = rs; }
    }

    // ---- normalize + coalesced float4 store
    for (int c = lane; c < 256; c += 32) {
      const int m = c >> 4, q = c & 15;
      const int r = row0 + m;
      if (r < total_nodes) {
        const float4 e  = *(const float4*)&S.tg[m][q * 4];
        const float  mu = S.mu[m], rs = S.rs[m];
        const float4 gm = ((const float4*)gamma_)[q];
        const float4 bt = ((const float4*)beta_)[q];
        float4 o;
        o.x = (e.x - mu) * rs * gm.x + bt.x;
        o.y = (e.y - mu) * rs * gm.y + bt.y;
        o.z = (e.z - mu) * rs * gm.z + bt.z;
        o.w = (e.w - mu) * rs * gm.w + bt.w;
        ((float4*)out)[(size_t)r * 16 + q] = o;
      }
    }
  }
}

extern "C" void kernel_launch(void* const* d_in, const int* in_sizes, int n_in,
                              void* d_out, int out_size, void* d_ws, size_t ws_size,
                              hipStream_t stream)
{
  const float* node_feat = (const float*)d_in[0];
  const float* text_feat = (const float*)d_in[1];
  const int*   seg_ids   = (const int*)d_in[2];
  const float* W1   = (const float*)d_in[3];
  const float* b1   = (const float*)d_in[4];
  const float* W2   = (const float*)d_in[5];
  const float* b2   = (const float*)d_in[6];
  const float* Wg   = (const float*)d_in[7];
  const float* bg   = (const float*)d_in[8];
  const float* qthr = (const float*)d_in[9];
  const float* gam  = (const float*)d_in[10];
  const float* bet  = (const float*)d_in[11];

  const int total_nodes = in_sizes[0] / 64;
  const int batch       = in_sizes[1] / 64;

  float* text_t  = (float*)d_ws;                       // batch*64 f32
  float* tgate   = text_t + (size_t)batch * 64;        // batch*64 f32
  float* quality = tgate  + (size_t)batch * 64;        // batch    f32

  text_kernel<<<batch, 64, 0, stream>>>(text_feat, W1, b1, W2, b2, Wg, bg, qthr,
                                        text_t, tgate, quality, batch);

  const int ntiles          = (total_nodes + 15) / 16;
  const int tiles_per_block = WAVES_PER_BLOCK * TILES_PER_WAVE;
  const int blocks          = (ntiles + tiles_per_block - 1) / tiles_per_block;
  fusion_kernel<<<blocks, WAVES_PER_BLOCK * 32, 0, stream>>>(
      node_feat, seg_ids, Wg, text_t, tgate, quality, gam, bet,
      (float*)d_out, total_nodes);
}
